// Seq2SeqModel_75093208203473
// MI455X (gfx1250) — matrix operand order repaired and checked by reading, not compiled, not run on hardware
//
#include <hip/hip_runtime.h>
#include <math.h>

// ---------------------------------------------------------------------------
// Seq2Seq leaky-RNN for MI455X (gfx1250, wave32, WMMA bf16 16x16x32)
//
// Problem: B=64, T=512, N_ENC=1024, N_DEC=512, IN_C=OUT_C=2, alpha=0.1
// Encoder scan (serial in T) dominates: 512 x (64x1024)@(1024x1024).
//
// Design:
//  * 64 persistent WGs = 4 row-groups (16 batch rows) x 16 col-tiles (64 cols)
//  * Each WG pins its J column slice (64x1024 bf16 = 128KB) in LDS once,
//    pre-swizzled into the exact B-fragment layout of v_wmma_f32_16x16x32_bf16.
//  * h lives in f32 accumulator registers (v8f per wave, D-tile layout).
//  * Per step: scatter tanh(h) (bf16, A-fragment layout) to a double-buffered
//    global exchange buffer -> grid barrier -> 32 WMMAs (K=1024) -> leaky
//    update fused with drive = noise + (x*s)@W_in.T -> Batch_T snapshot.
//  * Grid barrier: monotonic atomic counter (no reset race), 1 barrier/step.
// ---------------------------------------------------------------------------

typedef __attribute__((ext_vector_type(16))) __bf16 v16bf;
typedef __attribute__((ext_vector_type(8)))  float  v8f;

#define B_     64
#define T_     512
#define NENC   1024
#define NDEC   512
#define ALPHA_ 0.1f

#define NWG       64     // 4 rowgroups * 16 coltiles
#define WGTHREADS 128    // 4 wave32
#define NCOLS     64     // output cols per WG
#define KB        32     // 1024 / 32 k-blocks

__device__ __forceinline__ float fast_tanh(float x) {
#if __has_builtin(__builtin_amdgcn_tanhf)
  return __builtin_amdgcn_tanhf(x);      // v_tanh_f32 (gfx1250 TRANS op)
#elif __has_builtin(__builtin_amdgcn_tanh_f32)
  return __builtin_amdgcn_tanh_f32(x);
#else
  return tanhf(x);
#endif
}

__device__ __forceinline__ __bf16 to_bf16(float f) {
  unsigned u = __builtin_bit_cast(unsigned, f);
  u += 0x7FFFu + ((u >> 16) & 1u);                 // round-to-nearest-even
  unsigned short s = (unsigned short)(u >> 16);
  return __builtin_bit_cast(__bf16, s);
}

// Monotonic grid barrier: each round waits for counter >= target.
__device__ __forceinline__ void grid_barrier(unsigned* bar, unsigned target) {
  __syncthreads();
  if (threadIdx.x == 0) {
    __threadfence();                               // release stores (Abuf)
    atomicAdd(bar, 1u);
    while (*((volatile unsigned*)bar) < target) {
      __builtin_amdgcn_s_sleep(2);
    }
    __threadfence();                               // acquire for Abuf loads
  }
  __syncthreads();
}

// ---------------------------------------------------------------------------
// Persistent encoder scan kernel.
// Abuf layout: [buf(2)][rg(4)][kb(32)][lane(32)][16 bf16]   (A-fragment form)
// Jlds layout: [kb(32)][ntile(4)][lane(32)][16 bf16]        (B-fragment form)
// ---------------------------------------------------------------------------
__global__ void rnn_scan_kernel(const float* __restrict__ xin,      // (B,T,2)
                                const float* __restrict__ noise,    // (B,T,NENC)
                                const float* __restrict__ W_in,     // (NENC,2)
                                const float* __restrict__ J,        // (NENC,NENC)
                                const float* __restrict__ in_s,     // (2)
                                const int*   __restrict__ Batch_T,  // (B)
                                float*       __restrict__ last_h,   // (B,NENC), pre-zeroed
                                __bf16*      __restrict__ Abuf,     // 2*4*32*32*16 bf16
                                unsigned*    __restrict__ bar) {
  extern __shared__ __bf16 Jlds[];                 // 128 KB dynamic LDS

  const int wg    = blockIdx.x;
  const int rg    = wg >> 4;                       // row-group 0..3
  const int wgcol = wg & 15;                       // col-tile  0..15
  const int tid   = threadIdx.x;
  const int wave  = tid >> 5;                      // n-subtile 0..3
  const int lane  = tid & 31;
  const int hi    = lane >> 4;
  const int ln    = lane & 15;

  // ---- pack J[n0:n0+64, :] into LDS in B-fragment layout (once) ----
  // B frag (K=32 x N=16): lane: n = ln, koff = hi*8 + (p&7) + (p>>3)*16
  for (int i = tid; i < KB * 4 * 32 * 16; i += WGTHREADS) {
    int p   = i & 15;
    int l   = (i >> 4) & 31;
    int nt  = (i >> 9) & 3;
    int kb  = i >> 11;
    int n_g = wgcol * NCOLS + nt * 16 + (l & 15);
    int k   = kb * 32 + ((l >> 4) << 3) + (p & 7) + ((p >> 3) << 4);
    Jlds[((kb * 4 + nt) * 32 + l) * 16 + p] = to_bf16(J[(size_t)n_g * NENC + k]);
  }
  __syncthreads();

  // ---- per-lane constants ----
  const int   n_g = wgcol * NCOLS + wave * 16 + ln;   // output col == A k-index
  const float w0  = W_in[n_g * 2 + 0] * in_s[0];
  const float w1  = W_in[n_g * 2 + 1] * in_s[1];

  int bm[8], bt[8];
#pragma unroll
  for (int r = 0; r < 8; ++r) {
    bm[r] = rg * 16 + r + 8 * hi;                  // batch row of acc[r] (D layout)
    bt[r] = Batch_T[bm[r]];
  }

  // A-fragment scatter coords for k = n_g (ISA 16-bit A 16x32 layout)
  const int kb_s   = n_g >> 5;
  const int koff_s = n_g & 31;
  int laneA_base, pA;
  if      (koff_s <  8) { laneA_base = 0;  pA = koff_s;      }
  else if (koff_s < 16) { laneA_base = 16; pA = koff_s - 8;  }
  else if (koff_s < 24) { laneA_base = 0;  pA = koff_s - 8;  }
  else                  { laneA_base = 16; pA = koff_s - 16; }

  float hreg[8] = {0.f, 0.f, 0.f, 0.f, 0.f, 0.f, 0.f, 0.f};
  unsigned target = NWG;

  for (int t = 1; t <= T_; ++t) {
    const int buf = t & 1;
    __bf16* Ab = Abuf + (size_t)(buf * 4 + rg) * (KB * 32 * 16);

    // 1) publish A = tanh(h_{t-1}) in fragment layout (each (m,k) written once)
#pragma unroll
    for (int r = 0; r < 8; ++r) {
      int m = r + 8 * hi;
      Ab[(kb_s * 32 + (m + laneA_base)) * 16 + pA] = to_bf16(fast_tanh(hreg[r]));
    }

    // 2) one grid barrier per step (double-buffered Abuf)
    grid_barrier(bar, target);
    target += NWG;

    // 3) D = tanh(h) @ J.T for this 16x16 subtile, K = 1024
    v8f acc = {};
#pragma unroll 4
    for (int kb = 0; kb < KB; ++kb) {
      v16bf a = *(const v16bf*)(Ab   + ((size_t)(kb * 32 + lane)) * 16);
      v16bf b = *(const v16bf*)(Jlds + ((size_t)((kb * 4 + wave) * 32 + lane)) * 16);
      acc = __builtin_amdgcn_wmma_f32_16x16x32_bf16(
          false, a, false, b, (short)0, acc, false, false);
    }

    // 4) fused drive + leaky update + Batch_T snapshot
    const int td = t - 1;
#pragma unroll
    for (int r = 0; r < 8; ++r) {
      const float2 x = *(const float2*)(xin + ((size_t)bm[r] * T_ + td) * 2);
      const float  nv = noise[((size_t)bm[r] * T_ + td) * NENC + n_g];
      const float  drive = nv + w0 * x.x + w1 * x.y;
      hreg[r] = (1.0f - ALPHA_) * hreg[r] + ALPHA_ * (acc[r] + drive);
      if (bt[r] == t) last_h[bm[r] * NENC + n_g] = hreg[r];
    }
  }
}

// ---------------------------------------------------------------------------
// Tail (tiny): tanh gather state, W-projection, 3 decoder steps, output head.
// dec_traj written directly into d_out[0 .. 64*4*512), Output after it.
// ---------------------------------------------------------------------------
__global__ void tanh_inplace_kernel(float* __restrict__ h, int n) {
  int i = blockIdx.x * blockDim.x + threadIdx.x;
  if (i < n) h[i] = fast_tanh(h[i]);
}

__global__ void proj_kernel(const float* __restrict__ th,   // tanh(last_h) (B,NENC)
                            const float* __restrict__ W,    // (NDEC,NENC)
                            const float* __restrict__ bW,   // (NDEC)
                            float* __restrict__ dec) {      // d_out dec_traj
  int i = blockIdx.x * blockDim.x + threadIdx.x;            // 64*512 threads
  int b = i >> 9, n = i & 511;
  const float4* hv = (const float4*)(th + (size_t)b * NENC);
  const float4* wv = (const float4*)(W + (size_t)n * NENC);
  float acc = bW[n];
  for (int k = 0; k < NENC / 4; ++k) {
    float4 a = hv[k], w = wv[k];
    acc += a.x * w.x + a.y * w.y + a.z * w.z + a.w * w.w;
  }
  dec[(size_t)b * (4 * NDEC) + n] = acc;                    // dec_traj[b][0][n]
}

__global__ void dec_step_kernel(const float* __restrict__ Q,  // (NDEC,NDEC)
                                float* __restrict__ dec, int s) {
  int i = blockIdx.x * blockDim.x + threadIdx.x;            // 64*512 threads
  int b = i >> 9, n = i & 511;
  const float4* hv = (const float4*)(dec + (size_t)b * (4 * NDEC) + s * NDEC);
  const float4* qv = (const float4*)(Q + (size_t)n * NDEC);
  float acc = 0.f;
  for (int k = 0; k < NDEC / 4; ++k) {
    float4 a = hv[k], q = qv[k];
    acc += a.x * q.x + a.y * q.y + a.z * q.z + a.w * q.w;
  }
  dec[(size_t)b * (4 * NDEC) + (s + 1) * NDEC + n] = acc;
}

__global__ void out_kernel(const float* __restrict__ dec,
                           const float* __restrict__ W_out,   // (2,NDEC)
                           const float* __restrict__ out_s,   // (2)
                           float* __restrict__ out) {         // (B,4,2)
  int i = blockIdx.x * blockDim.x + threadIdx.x;              // 512 threads
  int b = i >> 3, s = (i >> 1) & 3, o = i & 1;
  const float* hv = dec + (size_t)b * (4 * NDEC) + s * NDEC;
  const float* wv = W_out + (size_t)o * NDEC;
  float acc = 0.f;
  for (int k = 0; k < NDEC; ++k) acc += hv[k] * wv[k];
  out[i] = out_s[o] * acc;
}

// ---------------------------------------------------------------------------
extern "C" void kernel_launch(void* const* d_in, const int* in_sizes, int n_in,
                              void* d_out, int out_size, void* d_ws, size_t ws_size,
                              hipStream_t stream) {
  const float* xin    = (const float*)d_in[0];   // Batch_Input (64,512,2)
  const float* noise  = (const float*)d_in[1];   // noise_rec   (64,512,1024)
  const float* W_in   = (const float*)d_in[2];   // (1024,2)
  const float* J      = (const float*)d_in[3];   // (1024,1024)
  const float* W      = (const float*)d_in[4];   // (512,1024)
  const float* bW     = (const float*)d_in[5];   // (512)
  const float* Q      = (const float*)d_in[6];   // (512,512)
  const float* W_out  = (const float*)d_in[7];   // (2,512)
  const float* in_s   = (const float*)d_in[8];   // (2)
  const float* out_s  = (const float*)d_in[9];   // (2)
  const int*   BatchT = (const int*)d_in[10];    // (64)
  // d_in[11] = decoder_steps == 3 (compile-time constant in reference)

  float* out = (float*)d_out;                    // [dec_traj 64*4*512 | Output 64*4*2]

  char*     ws     = (char*)d_ws;
  unsigned* bar    = (unsigned*)ws;                                  // 256 B
  float*    last_h = (float*)(ws + 256);                             // 256 KB
  __bf16*   Abuf   = (__bf16*)(ws + 256 + (size_t)B_ * NENC * 4);    // 256 KB

  // zero barrier counter + last_h (handles Batch_T==0 -> h0==0)
  hipMemsetAsync(ws, 0, 256 + (size_t)B_ * NENC * 4, stream);

  // encoder scan: 64 persistent WGs, 128 threads, 128KB dynamic LDS each
  rnn_scan_kernel<<<NWG, WGTHREADS, KB * 4 * 32 * 16 * sizeof(__bf16), stream>>>(
      xin, noise, W_in, J, in_s, BatchT, last_h, Abuf, bar);

  // last = tanh(last_h) @ W.T + b_W  -> dec_traj[:,0,:]
  tanh_inplace_kernel<<<(B_ * NENC + 255) / 256, 256, 0, stream>>>(last_h, B_ * NENC);
  proj_kernel<<<(B_ * NDEC) / 256, 256, 0, stream>>>(last_h, W, bW, out);

  // decoder: h <- h @ Q.T, 3 steps
  for (int s = 0; s < 3; ++s)
    dec_step_kernel<<<(B_ * NDEC) / 256, 256, 0, stream>>>(Q, out, s);

  // Output = out_strength * (dec_traj @ W_out.T)
  out_kernel<<<2, 256, 0, stream>>>(out, W_out, out_s, out + (size_t)B_ * 4 * NDEC);
}
